// EnhancedMamba2Mixer_66065186947611
// MI455X (gfx1250) — matrix operand order repaired and checked by reading, not compiled, not run on hardware
//
#include <hip/hip_runtime.h>
#include <hip/hip_bf16.h>

typedef __bf16 bf16_t;
typedef __attribute__((ext_vector_type(16))) __bf16 v16bf;
typedef __attribute__((ext_vector_type(8)))  float  v8f;

#define HID      2048
#define INTER    4096
#define NH       64
#define HD       64
#define NSTATE   128
#define KCONV    4
#define CONV_DIM (INTER + 2 * NSTATE)      // 4352
#define PROJ     (INTER + CONV_DIM + NH)   // 8512
#define PROJ_PAD 8576                      // 67 * 128 (padded N for 128-wide tiles)
#define BATCH    2
#define SEQ      2048
#define ROWS     (BATCH * SEQ)             // 4096
#define EPS_RMS  1e-5f

#define BM 128
#define BN 128
#define BK 32

// ---------------------------------------------------------------------------
// f32 -> bf16 conversion (grid-stride)
// ---------------------------------------------------------------------------
__global__ void cvt_f32_bf16(const float* __restrict__ in,
                             bf16_t* __restrict__ out, long n) {
    long i = (long)blockIdx.x * blockDim.x + threadIdx.x;
    long stride = (long)gridDim.x * blockDim.x;
    for (; i < n; i += stride) out[i] = (bf16_t)in[i];
}

// ---------------------------------------------------------------------------
// bf16 WMMA GEMM:  C[M, ldc] = A[M,K] * Bw[Npad,K]^T
// Block tile 128x128, BK=32, 256 threads = 8 waves, wave tile 32x64,
// double-buffered LDS (one barrier / K-step), 8 WMMAs / wave / K-step.
// ---------------------------------------------------------------------------
union FragBF { v16bf v; unsigned int u[8]; };

__launch_bounds__(256)
__global__ void gemm_bf16_wmma(const bf16_t* __restrict__ A,
                               const bf16_t* __restrict__ Bw,
                               float* __restrict__ C,
                               int ldc, int K) {
    __shared__ __align__(16) bf16_t As[2][BM * BK];
    __shared__ __align__(16) bf16_t Bs[2][BN * BK];

    const int t      = threadIdx.x;
    const int lane   = t & 31;
    const int wave   = t >> 5;
    const int wm     = wave & 3;        // 0..3 -> 32-row group
    const int wn     = wave >> 2;       // 0..1 -> 64-col group
    const int laneLo = lane & 15;
    const int laneHi = lane >> 4;
    const int mBase  = blockIdx.y * BM;
    const int nBase  = blockIdx.x * BN;

    v8f acc[2][4];
    const v8f vzero = {0.f, 0.f, 0.f, 0.f, 0.f, 0.f, 0.f, 0.f};
#pragma unroll
    for (int mi = 0; mi < 2; ++mi)
#pragma unroll
        for (int ni = 0; ni < 4; ++ni) acc[mi][ni] = vzero;

    // staging: 2 threads per row, 16 bf16 (2 x uint4) each
    const int sr = t >> 1, sc = (t & 1) * 16;
    const bf16_t* Aptr = A  + (size_t)(mBase + sr) * K + sc;
    const bf16_t* Bptr = Bw + (size_t)(nBase + sr) * K + sc;

    auto stage = [&](int buf, int k0) {
        const uint4* as = (const uint4*)(Aptr + k0);
        *(uint4*)&As[buf][sr * BK + sc]     = as[0];
        *(uint4*)&As[buf][sr * BK + sc + 8] = as[1];
        const uint4* bs = (const uint4*)(Bptr + k0);
        *(uint4*)&Bs[buf][sr * BK + sc]     = bs[0];
        *(uint4*)&Bs[buf][sr * BK + sc + 8] = bs[1];
    };

    stage(0, 0);
    __syncthreads();

    const int nk = K / BK;
    for (int kt = 0; kt < nk; ++kt) {
        const int cur = kt & 1;
        if (kt + 1 < nk) stage(cur ^ 1, (kt + 1) * BK);   // overlap with WMMA

        FragBF afrag[2], bfrag[4];
#pragma unroll
        for (int mi = 0; mi < 2; ++mi) {
            const int row = wm * 32 + mi * 16 + laneLo;
            // A fragment in this code's k order (same as B): lane holds one row, 16 contiguous K (lanes 16-31 +16)
#pragma unroll
            for (int p = 0; p < 8; ++p) {
                const int k = 2 * p + 16 * laneHi;
                afrag[mi].u[p] = *(const unsigned int*)&As[cur][row * BK + k];
            }
        }
#pragma unroll
        for (int ni = 0; ni < 4; ++ni) {
            const int col = wn * 64 + ni * 16 + laneLo;
            // B fragment in this code's k order (same as A): lane holds one column, 16 contiguous K
            const unsigned int* src =
                (const unsigned int*)&Bs[cur][col * BK + laneHi * 16];
#pragma unroll
            for (int p = 0; p < 8; ++p) bfrag[ni].u[p] = src[p];
        }

#pragma unroll
        for (int mi = 0; mi < 2; ++mi)
#pragma unroll
            for (int ni = 0; ni < 4; ++ni)
                acc[mi][ni] = __builtin_amdgcn_wmma_f32_16x16x32_bf16(
                    false, afrag[mi].v, false, bfrag[ni].v,
                    (short)0, acc[mi][ni], false, false);
        __syncthreads();
    }

    // C/D layout: VGPR r, lanes 0-15 -> M=r, lanes 16-31 -> M=r+8
#pragma unroll
    for (int mi = 0; mi < 2; ++mi)
#pragma unroll
        for (int ni = 0; ni < 4; ++ni) {
            const int row0 = mBase + wm * 32 + mi * 16 + 8 * laneHi;
            const int col  = nBase + wn * 64 + ni * 16 + laneLo;
#pragma unroll
            for (int r = 0; r < 8; ++r)
                C[(size_t)(row0 + r) * ldc + col] = acc[mi][ni][r];
        }
}

// ---------------------------------------------------------------------------
// causal depthwise conv1d (K=4) + bias + SiLU over proj[:, INTER:INTER+CONV_DIM]
// ---------------------------------------------------------------------------
__global__ void conv_silu(const float* __restrict__ proj,
                          const float* __restrict__ cw,
                          const float* __restrict__ cb,
                          float* __restrict__ hbc) {
    long i = (long)blockIdx.x * blockDim.x + threadIdx.x;
    const long total = (long)ROWS * CONV_DIM;
    const long stride = (long)gridDim.x * blockDim.x;
    for (; i < total; i += stride) {
        const int c   = (int)(i % CONV_DIM);
        const int row = (int)(i / CONV_DIM);     // b*SEQ + s
        const int s   = row % SEQ;
        float acc = cb[c];
        const float* w = cw + c * KCONV;
#pragma unroll
        for (int k = 0; k < KCONV; ++k) {
            const int sp = s + k - (KCONV - 1);
            if (sp >= 0)
                acc += w[k] *
                       proj[(size_t)(row + k - (KCONV - 1)) * PROJ_PAD + INTER + c];
        }
        hbc[i] = acc / (1.f + __expf(-acc));     // SiLU
    }
}

// ---------------------------------------------------------------------------
// dt = softplus(proj[..., INTER+CONV_DIM+h] + dt_bias[h])
// ---------------------------------------------------------------------------
__global__ void dt_softplus(const float* __restrict__ proj,
                            const float* __restrict__ dt_bias,
                            float* __restrict__ dts) {
    long i = (long)blockIdx.x * blockDim.x + threadIdx.x;
    const long total = (long)ROWS * NH;
    const long stride = (long)gridDim.x * blockDim.x;
    for (; i < total; i += stride) {
        const int h   = (int)(i % NH);
        const int row = (int)(i / NH);
        const float x =
            proj[(size_t)row * PROJ_PAD + INTER + CONV_DIM + h] + dt_bias[h];
        dts[i] = (x > 20.f) ? x : log1pf(__expf(x));
    }
}

// ---------------------------------------------------------------------------
// selective SSM scan: 1 block per (b,h); state 64x128 f32 in registers
// (32 floats/thread); double-buffered per-step staging (one barrier/step).
// thread t: d = t>>2, n-range = (t&3)*32 .. +32
// ---------------------------------------------------------------------------
__launch_bounds__(256)
__global__ void ssm_scan(const float* __restrict__ hbc,
                         const float* __restrict__ dts,
                         const float* __restrict__ A_log,
                         const float* __restrict__ Dp,
                         float* __restrict__ y) {
    const int blk = blockIdx.x;          // b*NH + h
    const int b = blk / NH, h = blk % NH;
    const int t = threadIdx.x;
    const int d = t >> 2, q = t & 3, n0 = q * 32;
    const float Ah = -__expf(A_log[h]);
    const float Dh = Dp[h];

    __shared__ float sB[2][NSTATE], sC[2][NSTATE], sX[2][HD];
    __shared__ float sdt[2];

    auto stage = [&](int buf, int s) {
        const size_t rowoff = (size_t)(b * SEQ + s) * CONV_DIM;
        if (t < NSTATE)          sB[buf][t]          = hbc[rowoff + INTER + t];
        else if (t < 2 * NSTATE) sC[buf][t - NSTATE] = hbc[rowoff + INTER + t];
        if (t < HD)              sX[buf][t]          = hbc[rowoff + h * HD + t];
        if (t == 255)            sdt[buf] = dts[(size_t)(b * SEQ + s) * NH + h];
    };

    float st[32];
#pragma unroll
    for (int i = 0; i < 32; ++i) st[i] = 0.f;

    stage(0, 0);
    __syncthreads();

    for (int s = 0; s < SEQ; ++s) {
        const int cur = s & 1;
        if (s + 1 < SEQ) stage(cur ^ 1, s + 1);   // overlap loads with compute

        const float dtv = sdt[cur];
        const float dA  = __expf(dtv * Ah);
        const float xd  = sX[cur][d];
        const float cx  = dtv * xd;
        float acc = 0.f;
#pragma unroll
        for (int i = 0; i < 32; ++i) {
            st[i] = st[i] * dA + cx * sB[cur][n0 + i];
            acc  += st[i] * sC[cur][n0 + i];
        }
        // reduce over the 4 lanes sharing d (contiguous lanes, wave32)
        acc += __shfl_xor(acc, 1, 32);
        acc += __shfl_xor(acc, 2, 32);
        if (q == 0)
            y[(size_t)(b * SEQ + s) * INTER + h * HD + d] = acc + Dh * xd;
        __syncthreads();
    }
}

// ---------------------------------------------------------------------------
// gated RMSNorm:  y *= silu(gate); y *= rsqrt(mean(y^2)+eps)*w; -> bf16
// 1 block per (b,s) row; 16 elems/thread
// ---------------------------------------------------------------------------
__launch_bounds__(256)
__global__ void gated_rmsnorm(const float* __restrict__ y,
                              const float* __restrict__ proj,
                              const float* __restrict__ nw,
                              bf16_t* __restrict__ out) {
    const int row = blockIdx.x;
    const int t   = threadIdx.x;
    const size_t ybase = (size_t)row * INTER;
    const size_t pbase = (size_t)row * PROJ_PAD;

    float v[16];
    float ss = 0.f;
#pragma unroll
    for (int j = 0; j < 16; ++j) {
        const int i = j * 256 + t;
        const float g  = proj[pbase + i];                 // gate (first INTER)
        const float sg = g / (1.f + __expf(-g));
        const float val = y[ybase + i] * sg;
        v[j] = val;
        ss += val * val;
    }
    __shared__ float red[256];
    red[t] = ss;
    __syncthreads();
    for (int off = 128; off > 0; off >>= 1) {
        if (t < off) red[t] += red[t + off];
        __syncthreads();
    }
    const float scale = rsqrtf(red[0] / (float)INTER + EPS_RMS);
#pragma unroll
    for (int j = 0; j < 16; ++j) {
        const int i = j * 256 + t;
        out[ybase + i] = (bf16_t)(v[j] * scale * nw[i]);
    }
}

// ---------------------------------------------------------------------------
// launch
// ---------------------------------------------------------------------------
extern "C" void kernel_launch(void* const* d_in, const int* in_sizes, int n_in,
                              void* d_out, int out_size, void* d_ws, size_t ws_size,
                              hipStream_t stream) {
    const float* hidden   = (const float*)d_in[0];   // [B,S,HID]
    const float* in_w     = (const float*)d_in[1];   // [PROJ,HID]
    const float* conv_w   = (const float*)d_in[2];   // [CONV_DIM,1,K]
    const float* conv_b   = (const float*)d_in[3];   // [CONV_DIM]
    const float* dt_bias  = (const float*)d_in[4];   // [NH]
    const float* A_log    = (const float*)d_in[5];   // [NH]
    const float* Dp       = (const float*)d_in[6];   // [NH]
    const float* norm_w   = (const float*)d_in[7];   // [INTER]
    const float* out_w    = (const float*)d_in[8];   // [HID,INTER]
    float* out = (float*)d_out;                      // [B,S,HID]

    // workspace partition (256B aligned)
    size_t off = 0;
    char* ws = (char*)d_ws;
    auto take = [&](size_t bytes) -> void* {
        void* p = ws + off;
        off += (bytes + 255) & ~(size_t)255;
        return p;
    };
    bf16_t* Xbf  = (bf16_t*)take((size_t)ROWS * HID      * sizeof(bf16_t));
    bf16_t* Wbf  = (bf16_t*)take((size_t)PROJ_PAD * HID  * sizeof(bf16_t)); // padded rows
    bf16_t* OWbf = (bf16_t*)take((size_t)HID  * INTER    * sizeof(bf16_t));
    float*  proj = (float*) take((size_t)ROWS * PROJ_PAD * sizeof(float));  // padded cols
    float*  hbc  = (float*) take((size_t)ROWS * CONV_DIM * sizeof(float));
    float*  dts  = (float*) take((size_t)ROWS * NH       * sizeof(float));
    float*  yb   = (float*) take((size_t)ROWS * INTER    * sizeof(float));
    bf16_t* Ybf  = (bf16_t*)take((size_t)ROWS * INTER    * sizeof(bf16_t));
    (void)ws_size; (void)n_in; (void)in_sizes; (void)out_size;

    // 1) convert to bf16 (padded tail rows of Wbf stay garbage: their products
    //    land only in padded proj columns, which no consumer reads)
    cvt_f32_bf16<<<2048, 256, 0, stream>>>(hidden, Xbf,  (long)ROWS * HID);
    cvt_f32_bf16<<<4096, 256, 0, stream>>>(in_w,   Wbf,  (long)PROJ * HID);
    cvt_f32_bf16<<<2048, 256, 0, stream>>>(out_w,  OWbf, (long)HID  * INTER);

    // 2) in_proj GEMM: proj[ROWS, PROJ_PAD] = X * W^T
    {
        dim3 grid(PROJ_PAD / BN, ROWS / BM);      // 67 x 32
        gemm_bf16_wmma<<<grid, 256, 0, stream>>>(Xbf, Wbf, proj,
                                                 PROJ_PAD, HID);
    }

    // 3) causal depthwise conv + SiLU
    conv_silu<<<8192, 256, 0, stream>>>(proj, conv_w, conv_b, hbc);

    // 4) dt softplus
    dt_softplus<<<1024, 256, 0, stream>>>(proj, dt_bias, dts);

    // 5) selective scan (128 blocks = B*NH)
    ssm_scan<<<BATCH * NH, 256, 0, stream>>>(hbc, dts, A_log, Dp, yb);

    // 6) gated RMSNorm -> bf16
    gated_rmsnorm<<<ROWS, 256, 0, stream>>>(yb, proj, norm_w, Ybf);

    // 7) out_proj GEMM: out[ROWS, HID] = Ybf * OW^T
    {
        dim3 grid(HID / BN, ROWS / BM);           // 16 x 32
        gemm_bf16_wmma<<<grid, 256, 0, stream>>>(Ybf, OWbf, out,
                                                 HID, INTER);
    }
}
